// TransformerBlock_5239860101720
// MI455X (gfx1250) — matrix operand order repaired and checked by reading, not compiled, not run on hardware
//
#include <hip/hip_runtime.h>
#include <hip/hip_bf16.h>
#include <math.h>

// ---------------------------------------------------------------------------
// MI455X / gfx1250 transformer block. bf16 WMMA (16x16x32) with fp32 accum,
// async global->LDS double-buffered GEMM staging, flash attention.
// ---------------------------------------------------------------------------

typedef __attribute__((ext_vector_type(16))) __bf16 v16bf;
typedef __attribute__((ext_vector_type(8)))  float  v8f;
typedef __attribute__((ext_vector_type(8)))  unsigned short ushort8;
typedef __attribute__((ext_vector_type(4)))  int    v4i;
typedef unsigned short u16;

#define AS1 __attribute__((address_space(1)))
#define AS3 __attribute__((address_space(3)))

#if defined(__has_builtin)
#if __has_builtin(__builtin_amdgcn_global_load_async_to_lds_b128)
#define HAVE_ASYNC_LDS 1
#endif
#endif
#ifndef HAVE_ASYNC_LDS
#define HAVE_ASYNC_LDS 0
#endif

#if defined(__has_builtin)
#if __has_builtin(__builtin_amdgcn_s_wait_asynccnt)
#define WAIT_ASYNC(n) __builtin_amdgcn_s_wait_asynccnt(n)
#endif
#endif
#ifndef WAIT_ASYNC
#define WAIT_ASYNC(n) asm volatile("s_wait_asynccnt %0" ::"i"(n) : "memory")
#endif

#define D_MODEL 1024
#define N_HEADS 16
#define D_HEAD  64
#define D_FF    4096
#define SEQ_B   4
#define SEQ_L   2048
#define TOKENS  (SEQ_B * SEQ_L)   // 8192

static __device__ __forceinline__ u16 f2bf(float f) {
  unsigned int u = __builtin_bit_cast(unsigned int, f);
  u += 0x7fffu + ((u >> 16) & 1u);           // round-to-nearest-even
  return (u16)(u >> 16);
}
static __device__ __forceinline__ __bf16 asbf(u16 v) {
  return __builtin_bit_cast(__bf16, v);
}

static __device__ __forceinline__ v8f wmma_bf16(v16bf a, v16bf b, v8f c) {
  return __builtin_amdgcn_wmma_f32_16x16x32_bf16(false, a, false, b, (short)0, c,
                                                 false, false);
}

// --- Fragment loaders, CDNA5 ISA 7.12.2 layouts ---------------------------
// A fragment 16(M) x 32(K) from row-major [M][ld]:
//   lanes 0-15 : row=lane,    K 0-7 / 16-23 ;  lanes 16-31: K 8-15 / 24-31
// -> two contiguous 8x-u16 (128-bit) reads per lane.
static __device__ __forceinline__ v16bf frag_a(const u16* base, size_t ld, int lane) {
  const int row = lane & 15;
  const int kb  = (lane >> 4) << 3;
  const u16* p = base + (size_t)row * ld + kb;
  v16bf a;
#pragma unroll
  for (int i = 0; i < 4; ++i) {
    a[2*i]     = asbf(p[2*i]);
    a[2*i+1]   = asbf(p[2*i+1]);
    a[8+2*i]   = asbf(p[16+2*i]);
    a[8+2*i+1] = asbf(p[16+2*i+1]);
  }
  return a;
}
// B fragment 32(K) x 16(N) from [N][ld] storage (contraction contiguous):
//   lanes 0-15: col=lane, K 0-7 / 16-23 ; lanes 16-31: col=lane-16, K 8-15 / 24-31
// -> two contiguous 8x-u16 (128-bit) reads per lane.
static __device__ __forceinline__ v16bf frag_b_t(const u16* base, size_t ld, int lane) {
  const int col = lane & 15;
  const int kb  = (lane >> 4) << 3;
  const u16* p = base + (size_t)col * ld + kb;
  v16bf b;
#pragma unroll
  for (int i = 0; i < 4; ++i) {
    b[2*i]     = asbf(p[2*i]);
    b[2*i+1]   = asbf(p[2*i+1]);
    b[8+2*i]   = asbf(p[16+2*i]);
    b[8+2*i+1] = asbf(p[16+2*i+1]);
  }
  return b;
}

// ---------------------------------------------------------------------------
// Weight convert: fp32 [K][N] -> bf16 [N][K] (transpose so GEMM B-tiles are
// K-contiguous). One-time cost, coalesced writes.
// ---------------------------------------------------------------------------
__global__ __launch_bounds__(256) void convert_transpose(
    const float* __restrict__ in, u16* __restrict__ out, int K, int N) {
  int idx = blockIdx.x * 256 + threadIdx.x;
  if (idx >= N * K) return;
  int n = idx / K, k = idx - n * K;
  out[idx] = f2bf(in[(size_t)k * N + n]);
}

// V transpose for attention: qkv[t][2048+d] -> vT[d][t] (coalesced writes).
__global__ __launch_bounds__(256) void transpose_v(const u16* __restrict__ qkv,
                                                   u16* __restrict__ vT) {
  int idx = blockIdx.x * 256 + threadIdx.x;   // d * TOKENS + t
  int d = idx >> 13;                          // / 8192
  int t = idx & (TOKENS - 1);
  vT[idx] = qkv[(size_t)t * (3 * D_MODEL) + 2 * D_MODEL + d];
}

// ---------------------------------------------------------------------------
// LayerNorm (D=1024), one row per 256-thread block, bf16 out.
// ---------------------------------------------------------------------------
__global__ __launch_bounds__(256) void layernorm_to_bf16(
    const float* __restrict__ x, const float* __restrict__ g,
    const float* __restrict__ b, u16* __restrict__ out) {
  const int row = blockIdx.x;
  const int tid = threadIdx.x;
  const float* xr = x + (size_t)row * D_MODEL;
  float vals[4];
  float s = 0.f, s2 = 0.f;
#pragma unroll
  for (int i = 0; i < 4; ++i) {
    float v = xr[tid + i * 256];
    vals[i] = v; s += v; s2 += v * v;
  }
#pragma unroll
  for (int off = 16; off > 0; off >>= 1) {
    s  += __shfl_xor(s, off, 32);
    s2 += __shfl_xor(s2, off, 32);
  }
  __shared__ float red[2][8];
  const int wave = tid >> 5, lane = tid & 31;
  if (lane == 0) { red[0][wave] = s; red[1][wave] = s2; }
  __syncthreads();
  float ts = 0.f, ts2 = 0.f;
#pragma unroll
  for (int w = 0; w < 8; ++w) { ts += red[0][w]; ts2 += red[1][w]; }
  const float mu   = ts * (1.0f / D_MODEL);
  const float var  = ts2 * (1.0f / D_MODEL) - mu * mu;
  const float rstd = rsqrtf(var + 1e-5f);
#pragma unroll
  for (int i = 0; i < 4; ++i) {
    const int c = tid + i * 256;
    out[(size_t)row * D_MODEL + c] = f2bf((vals[i] - mu) * rstd * g[c] + b[c]);
  }
}

// ---------------------------------------------------------------------------
// Tiled bf16 WMMA GEMM: out = epi(A[MxK] @ Bt[NxK]^T + bias).
// 256 thr = 8 waves; 128x128 block tile, K-step 32; waves 4(M) x 2(N),
// each wave 32x64 = 2x4 WMMA frags. LDS tiles double-buffered via
// global_load_async_to_lds_b128 + s_wait_asynccnt (sync b128 fallback).
// ---------------------------------------------------------------------------
#define TM 128
#define TN 128
#define TK 32
#define LDT (TK + 8)   // 40 u16 rows (80B, 16B-aligned)

template <bool GELU, bool RESID, bool OUTBF>
__global__ __launch_bounds__(256) void gemm_bf16_wmma(
    const u16* __restrict__ A, const u16* __restrict__ Bt,
    const float* __restrict__ bias, const float* __restrict__ resid,
    void* __restrict__ outv, int M, int N, int K) {
  __shared__ u16 As[2][TM][LDT];
  __shared__ u16 Bs[2][TN][LDT];
  const int tid  = threadIdx.x;
  const int lane = tid & 31;
  const int wave = tid >> 5;
  const int wm = wave & 3;
  const int wn = wave >> 2;
  const int bm = blockIdx.y * TM;
  const int bn = blockIdx.x * TN;

  // Stage mapping: thread t covers tile row (t>>1), 16 u16 at col (t&1)*16.
  const int r = tid >> 1, c = (tid & 1) * 16;
  const u16* agsrc = A  + (size_t)(bm + r) * K + c;
  const u16* bgsrc = Bt + (size_t)(bn + r) * K + c;

  v8f acc[2][4];
#pragma unroll
  for (int i = 0; i < 2; ++i)
#pragma unroll
    for (int j = 0; j < 4; ++j)
      acc[i][j] = (v8f){0.f, 0.f, 0.f, 0.f, 0.f, 0.f, 0.f, 0.f};

  auto stage = [&](int buf, int k0) {
#if HAVE_ASYNC_LDS
    __builtin_amdgcn_global_load_async_to_lds_b128(
        (AS1 v4i*)(agsrc + k0),     (AS3 v4i*)&As[buf][r][c],     0, 0);
    __builtin_amdgcn_global_load_async_to_lds_b128(
        (AS1 v4i*)(agsrc + k0 + 8), (AS3 v4i*)&As[buf][r][c + 8], 0, 0);
    __builtin_amdgcn_global_load_async_to_lds_b128(
        (AS1 v4i*)(bgsrc + k0),     (AS3 v4i*)&Bs[buf][r][c],     0, 0);
    __builtin_amdgcn_global_load_async_to_lds_b128(
        (AS1 v4i*)(bgsrc + k0 + 8), (AS3 v4i*)&Bs[buf][r][c + 8], 0, 0);
#else
    const ushort8* asv = (const ushort8*)(agsrc + k0);
    ushort8* adv = (ushort8*)&As[buf][r][c];
    adv[0] = asv[0]; adv[1] = asv[1];
    const ushort8* bsv = (const ushort8*)(bgsrc + k0);
    ushort8* bdv = (ushort8*)&Bs[buf][r][c];
    bdv[0] = bsv[0]; bdv[1] = bsv[1];
#endif
  };

  const int nk = K / TK;
  stage(0, 0);
  for (int ki = 0; ki < nk; ++ki) {
    const int cur = ki & 1;
    const bool pre = (ki + 1) < nk;
    if (pre) stage(cur ^ 1, (ki + 1) * TK);
#if HAVE_ASYNC_LDS
    if (pre) { WAIT_ASYNC(4); } else { WAIT_ASYNC(0); }
#endif
    __syncthreads();

    v16bf af[2], bfr[4];
#pragma unroll
    for (int i = 0; i < 2; ++i)
      af[i] = frag_a(&As[cur][wm * 32 + i * 16][0], LDT, lane);
#pragma unroll
    for (int j = 0; j < 4; ++j)
      bfr[j] = frag_b_t(&Bs[cur][wn * 64 + j * 16][0], LDT, lane);
#pragma unroll
    for (int i = 0; i < 2; ++i)
#pragma unroll
      for (int j = 0; j < 4; ++j)
        acc[i][j] = wmma_bf16(af[i], bfr[j], acc[i][j]);
    __syncthreads();
  }

  // Epilogue. C-frag: col = lane&15, row = vgpr + 8*(lane>=16).
  const int colin = lane & 15;
  const int rowh  = (lane >> 4) << 3;
#pragma unroll
  for (int i = 0; i < 2; ++i) {
#pragma unroll
    for (int j = 0; j < 4; ++j) {
      const int gcol = bn + wn * 64 + j * 16 + colin;
      const float bv = bias[gcol];
#pragma unroll
      for (int vg = 0; vg < 8; ++vg) {
        const int grow = bm + wm * 32 + i * 16 + rowh + vg;
        float v = acc[i][j][vg] + bv;
        if (GELU) v = 0.5f * v * (1.0f + erff(v * 0.70710678118f));
        if (RESID) v += resid[(size_t)grow * N + gcol];
        if (OUTBF) ((u16*)outv)[(size_t)grow * N + gcol] = f2bf(v);
        else       ((float*)outv)[(size_t)grow * N + gcol] = v;
      }
    }
  }
}

// ---------------------------------------------------------------------------
// Flash attention (causal). qkv: [TOKENS][3072] bf16, vT: [1024][TOKENS] bf16.
// Each wave owns 16 queries, streams keys in 32-chunks with online softmax;
// QK^T and P@V on the WMMA pipe. Block = 8 waves = 128 queries.
// ---------------------------------------------------------------------------
__global__ __launch_bounds__(256) void attn_wmma(const u16* __restrict__ qkv,
                                                 const u16* __restrict__ vT,
                                                 u16* __restrict__ obf) {
  __shared__ u16 Plds[8][16][40];   // per-wave P re-layout (C-frag -> A-frag)
  const int tid = threadIdx.x, lane = tid & 31, wave = tid >> 5;
  const int lrow = lane & 15, lhalf = lane >> 4;
  const int bh = blockIdx.x;
  const int batch = bh >> 4, head = bh & 15;
  const int q0 = blockIdx.y * 128 + wave * 16;
  const size_t tok0 = (size_t)batch * SEQ_L;
  const size_t LDQ = 3 * D_MODEL;
  const u16* qp = qkv + tok0 * LDQ + head * D_HEAD;
  const u16* kp = qp + D_MODEL;
  const u16* vtp = vT + (size_t)head * D_HEAD * TOKENS + tok0;  // [d][key]

  v16bf qf[2];
#pragma unroll
  for (int d = 0; d < 2; ++d)
    qf[d] = frag_a(qp + (size_t)q0 * LDQ + d * 32, LDQ, lane);

  v8f oacc[4];
#pragma unroll
  for (int n = 0; n < 4; ++n)
    oacc[n] = (v8f){0.f, 0.f, 0.f, 0.f, 0.f, 0.f, 0.f, 0.f};
  float m_r[8], l_r[8];
#pragma unroll
  for (int vg = 0; vg < 8; ++vg) { m_r[vg] = -3.0e38f; l_r[vg] = 0.f; }

  const int kend = q0 + 16;
  for (int kc = 0; kc < kend; kc += 32) {
    if (kc + 32 < kend) {   // prefetch next K/V chunk -> global_prefetch_b8
      __builtin_prefetch(kp + (size_t)(kc + 32) * LDQ, 0, 1);
      __builtin_prefetch(vtp + kc + 32, 0, 1);
    }
    // S = (Q @ K^T) / 8, two 16-key subtiles
    v8f st[2];
#pragma unroll
    for (int t = 0; t < 2; ++t) {
      const int kt = kc + t * 16;
      v8f s = (v8f){0.f, 0.f, 0.f, 0.f, 0.f, 0.f, 0.f, 0.f};
#pragma unroll
      for (int d = 0; d < 2; ++d) {
        v16bf kf = frag_b_t(kp + (size_t)kt * LDQ + d * 32, LDQ, lane);
        s = wmma_bf16(qf[d], kf, s);
      }
      const int key = kt + lrow;
#pragma unroll
      for (int vg = 0; vg < 8; ++vg) {
        const int qr = q0 + vg + 8 * lhalf;
        s[vg] = (key <= qr) ? s[vg] * 0.125f : -3.0e38f;
      }
      st[t] = s;
    }
    // Online softmax per row (rows live in a 16-lane half -> xor<=8)
    float corr[8], p0[8], p1[8];
#pragma unroll
    for (int vg = 0; vg < 8; ++vg) {
      float cm = fmaxf(st[0][vg], st[1][vg]);
#pragma unroll
      for (int off = 8; off > 0; off >>= 1) cm = fmaxf(cm, __shfl_xor(cm, off, 32));
      const float mn = fmaxf(m_r[vg], cm);
      const float c  = __expf(m_r[vg] - mn);
      const float e0 = __expf(st[0][vg] - mn);
      const float e1 = __expf(st[1][vg] - mn);
      float rs = e0 + e1;
#pragma unroll
      for (int off = 8; off > 0; off >>= 1) rs += __shfl_xor(rs, off, 32);
      l_r[vg] = l_r[vg] * c + rs;
      m_r[vg] = mn;
      corr[vg] = c; p0[vg] = e0; p1[vg] = e1;
    }
#pragma unroll
    for (int n = 0; n < 4; ++n)
#pragma unroll
      for (int vg = 0; vg < 8; ++vg) oacc[n][vg] *= corr[vg];

    // P: C-frag -> A-frag via per-wave LDS (same-wave DS ops are in-order)
#pragma unroll
    for (int vg = 0; vg < 8; ++vg) {
      Plds[wave][vg + 8 * lhalf][lrow]      = f2bf(p0[vg]);
      Plds[wave][vg + 8 * lhalf][16 + lrow] = f2bf(p1[vg]);
    }
    asm volatile("s_wait_dscnt 0" ::: "memory");
    v16bf pf = frag_a(&Plds[wave][0][0], 40, lane);

    // O += P(16x32) @ V(32x64): V^T rows are key-contiguous -> b128 reads
#pragma unroll
    for (int n = 0; n < 4; ++n) {
      v16bf vf = frag_b_t(vtp + (size_t)(n * 16) * TOKENS + kc, TOKENS, lane);
      oacc[n] = wmma_bf16(pf, vf, oacc[n]);
    }
  }

#pragma unroll
  for (int n = 0; n < 4; ++n)
#pragma unroll
    for (int vg = 0; vg < 8; ++vg) {
      const int qr = q0 + vg + 8 * lhalf;
      const float v = oacc[n][vg] / l_r[vg];
      obf[(tok0 + qr) * D_MODEL + head * D_HEAD + n * 16 + lrow] = f2bf(v);
    }
}

// ---------------------------------------------------------------------------
// Host orchestration
// ---------------------------------------------------------------------------
extern "C" void kernel_launch(void* const* d_in, const int* in_sizes, int n_in,
                              void* d_out, int out_size, void* d_ws, size_t ws_size,
                              hipStream_t stream) {
  (void)in_sizes; (void)n_in; (void)out_size; (void)ws_size;
  const float* x    = (const float*)d_in[0];
  const float* ln1g = (const float*)d_in[1];
  const float* ln1b = (const float*)d_in[2];
  const float* wqkv = (const float*)d_in[3];
  const float* bqkv = (const float*)d_in[4];
  const float* wout = (const float*)d_in[5];
  const float* bout = (const float*)d_in[6];
  const float* ln2g = (const float*)d_in[7];
  const float* ln2b = (const float*)d_in[8];
  const float* w1   = (const float*)d_in[9];
  const float* b1   = (const float*)d_in[10];
  const float* w2   = (const float*)d_in[11];
  const float* b2   = (const float*)d_in[12];

  char* ws = (char*)d_ws;
  size_t off = 0;
  auto take = [&](size_t bytes) -> char* {
    char* p = ws + off;
    off += (bytes + 255) & ~(size_t)255;
    return p;
  };
  u16*   wqkv_bf = (u16*)take((size_t)D_MODEL * 3 * D_MODEL * 2);  // [3072][1024]
  u16*   wout_bf = (u16*)take((size_t)D_MODEL * D_MODEL * 2);      // [1024][1024]
  u16*   w1_bf   = (u16*)take((size_t)D_MODEL * D_FF * 2);         // [4096][1024]
  u16*   w2_bf   = (u16*)take((size_t)D_FF * D_MODEL * 2);         // [1024][4096]
  u16*   regA    = (u16*)take((size_t)TOKENS * D_MODEL * 2);       // xln / attn_o
  u16*   regB    = (u16*)take((size_t)TOKENS * 3 * D_MODEL * 2);   // qkv
  float* hbuf    = (float*)take((size_t)TOKENS * D_MODEL * 4);     // h (fp32)
  u16*   h2ln    = (u16*)take((size_t)TOKENS * D_MODEL * 2);       // LN2 out
  u16*   vT      = h2ln;  // V^T [1024][8192] aliases LN2 buf (disjoint lifetime)
  u16*   ffbf    = regA;  // FF act (67MB) aliases dead regA+regB

  dim3 blk(256);

  // 1) weights fp32 [K][N] -> bf16 [N][K]
  convert_transpose<<<(D_MODEL * 3 * D_MODEL + 255) / 256, blk, 0, stream>>>(
      wqkv, wqkv_bf, D_MODEL, 3 * D_MODEL);
  convert_transpose<<<(D_MODEL * D_MODEL + 255) / 256, blk, 0, stream>>>(
      wout, wout_bf, D_MODEL, D_MODEL);
  convert_transpose<<<(D_MODEL * D_FF + 255) / 256, blk, 0, stream>>>(
      w1, w1_bf, D_MODEL, D_FF);
  convert_transpose<<<(D_FF * D_MODEL + 255) / 256, blk, 0, stream>>>(
      w2, w2_bf, D_FF, D_MODEL);

  // 2) LN1(x) -> bf16
  layernorm_to_bf16<<<TOKENS, blk, 0, stream>>>(x, ln1g, ln1b, regA);

  // 3) QKV = xln @ w_qkv + b_qkv -> bf16 [TOKENS][3072]
  gemm_bf16_wmma<false, false, true><<<dim3(3 * D_MODEL / TN, TOKENS / TM), blk, 0, stream>>>(
      regA, wqkv_bf, bqkv, nullptr, regB, TOKENS, 3 * D_MODEL, D_MODEL);

  // 3.5) V^T for the AV matmul
  transpose_v<<<(TOKENS * D_MODEL) / 256, blk, 0, stream>>>(regB, vT);

  // 4) causal flash attention -> bf16 [TOKENS][1024]
  attn_wmma<<<dim3(SEQ_B * N_HEADS, SEQ_L / 128), blk, 0, stream>>>(regB, vT, regA);

  // 5) h = attn_o @ w_out + b_out + x  (fp32)
  gemm_bf16_wmma<false, true, false><<<dim3(D_MODEL / TN, TOKENS / TM), blk, 0, stream>>>(
      regA, wout_bf, bout, x, hbuf, TOKENS, D_MODEL, D_MODEL);

  // 6) LN2(h) -> bf16
  layernorm_to_bf16<<<TOKENS, blk, 0, stream>>>(hbuf, ln2g, ln2b, h2ln);

  // 7) ff = gelu(h2 @ w1 + b1) -> bf16 [TOKENS][4096]
  gemm_bf16_wmma<true, false, true><<<dim3(D_FF / TN, TOKENS / TM), blk, 0, stream>>>(
      h2ln, w1_bf, b1, nullptr, ffbf, TOKENS, D_FF, D_MODEL);

  // 8) out = ff @ w2 + b2 + h  (fp32, final)
  gemm_bf16_wmma<false, true, false><<<dim3(D_MODEL / TN, TOKENS / TM), blk, 0, stream>>>(
      ffbf, w2_bf, b2, hbuf, (float*)d_out, TOKENS, D_MODEL, D_FF);
}
